// Seq2Seq_86208583565521
// MI455X (gfx1250) — matrix operand order repaired and checked by reading, not compiled, not run on hardware
//
#include <hip/hip_runtime.h>

typedef __bf16 bf16;
typedef __attribute__((ext_vector_type(4)))  bf16  v4bf;
typedef __attribute__((ext_vector_type(8)))  bf16  v8bf;
typedef __attribute__((ext_vector_type(16))) bf16  v16bf;
typedef __attribute__((ext_vector_type(8)))  float v8f;

constexpr int B_   = 64;
constexpr int SEN  = 64;
constexpr int SFR  = 64;
constexpr int EMBD = 512;
constexpr int UN   = 1024;
constexpr int G4   = 4096;    // 4*UNITS
constexpr int VFR  = 16000;

// ---------------------------------------------------------------------------
// Multi-segment tiled BF16 WMMA GEMM: C(MxN) = sum_s A_s(MxK_s) @ B_s(K_s x N) [+bias]
// Workgroup tile 64x128, 8 waves, each wave 2x2 tiles of 16x16, K step 32.
// Requires M%64==0, N%128==0, K_s%32==0, 8B-aligned pointers/strides.
// ---------------------------------------------------------------------------
__launch_bounds__(256)
__global__ void gemm_bf16_wmma(
    const bf16* __restrict__ A0, long lda0, const bf16* __restrict__ B0, long ldb0, int K0,
    const bf16* __restrict__ A1, long lda1, const bf16* __restrict__ B1, long ldb1, int K1,
    const bf16* __restrict__ A2, long lda2, const bf16* __restrict__ B2, long ldb2, int K2,
    float* __restrict__ C, long ldc, const float* __restrict__ bias)
{
    __shared__ __align__(32) bf16 As2[64 * 32];    // A tile [row][k], natural k order
    __shared__ __align__(32) bf16 Bst[128 * 32];   // transposed B tile [col][k]

    const int tid  = threadIdx.x;
    const int lane = tid & 31;
    const int wid  = tid >> 5;
    const int wm   = wid >> 2;       // 0..1 : 32-row slab
    const int wn   = wid & 3;        // 0..3 : 32-col slab
    const int l    = lane & 15;
    const int hi   = lane >> 4;

    const long row0 = (long)blockIdx.y * 64;
    const long col0 = (long)blockIdx.x * 128;

    v8f acc00 = {0,0,0,0,0,0,0,0};
    v8f acc01 = acc00, acc10 = acc00, acc11 = acc00;

    const bf16* Aseg[3] = {A0, A1, A2};
    const bf16* Bseg[3] = {B0, B1, B2};
    const long  ldaS[3] = {lda0, lda1, lda2};
    const long  ldbS[3] = {ldb0, ldb1, ldb2};
    const int   Kseg[3] = {K0, K1, K2};

    // LDS staging roles
    const int ar   = tid >> 2;                                 // A row 0..63
    const int am   = tid & 3;                                  // A 8-col chunk
    const int adst = am * 8;                                   // natural k order
    const int bkq  = (tid >> 5) * 4;                           // B k-quad base (0..28)
    const int bcq  = (tid & 31) * 4;                           // B col-quad base (0..124)

    #pragma unroll
    for (int s = 0; s < 3; ++s) {
        const bf16* A = Aseg[s];
        const bf16* B = Bseg[s];
        const int   K = Kseg[s];
        if (A == nullptr || K == 0) continue;
        const long lda = ldaS[s], ldb = ldbS[s];
        for (int kk = 0; kk < K; kk += 32) {
            __syncthreads();
            // A 64x32 -> LDS, natural k order, so a lane's v16bf fragment is contiguous
            const bf16* ap = A + (row0 + ar) * lda + kk + am * 8;
            v8bf av = *(const v8bf*)ap;
            *(v8bf*)&As2[ar * 32 + adst] = av;
            // B 32x128 -> LDS transposed [col][k]; 4x4 micro-block per thread:
            // 4 x global_load_b64 in, 4 x ds_store_b64 out (no scalar b16 traffic)
            const bf16* bp = B + (long)(kk + bkq) * ldb + col0 + bcq;
            v4bf r0 = *(const v4bf*)(bp);
            v4bf r1 = *(const v4bf*)(bp + ldb);
            v4bf r2 = *(const v4bf*)(bp + 2 * ldb);
            v4bf r3 = *(const v4bf*)(bp + 3 * ldb);
            #pragma unroll
            for (int cc = 0; cc < 4; ++cc) {
                v4bf colv = { r0[cc], r1[cc], r2[cc], r3[cc] };
                *(v4bf*)&Bst[(bcq + cc) * 32 + bkq] = colv;
            }
            // Pull next K-tile toward the WGP while this one is consumed
            if (kk + 32 < K) {
                __builtin_prefetch((const void*)(ap + 32), 0, 1);
                __builtin_prefetch((const void*)(bp + 32 * ldb), 0, 1);
            }
            __syncthreads();

            // Fragments (k = 16*hi + s in A and B; B via transposed store)
            v16bf a0 = *(const v16bf*)&As2[(wm * 32 +      l) * 32 + hi * 16];
            v16bf a1 = *(const v16bf*)&As2[(wm * 32 + 16 + l) * 32 + hi * 16];
            v16bf b0 = *(const v16bf*)&Bst[(wn * 32 +      l) * 32 + hi * 16];
            v16bf b1 = *(const v16bf*)&Bst[(wn * 32 + 16 + l) * 32 + hi * 16];

            acc00 = __builtin_amdgcn_wmma_f32_16x16x32_bf16(false, a0, false, b0, (short)0, acc00, false, false);
            acc01 = __builtin_amdgcn_wmma_f32_16x16x32_bf16(false, a0, false, b1, (short)0, acc01, false, false);
            acc10 = __builtin_amdgcn_wmma_f32_16x16x32_bf16(false, a1, false, b0, (short)0, acc10, false, false);
            acc11 = __builtin_amdgcn_wmma_f32_16x16x32_bf16(false, a1, false, b1, (short)0, acc11, false, false);
        }
    }

    // Store: C/D layout -> M = hi*8 + e, N = lane&15 within each 16x16 tile
    const long cA = col0 + wn * 32 + l;
    const long cB = cA + 16;
    const long rb = row0 + wm * 32 + hi * 8;
    #pragma unroll
    for (int e = 0; e < 8; ++e) {
        const long r0 = rb + e, r1 = rb + 16 + e;
        const float b0v = bias ? bias[cA] : 0.f;
        const float b1v = bias ? bias[cB] : 0.f;
        C[r0 * ldc + cA] = acc00[e] + b0v;
        C[r0 * ldc + cB] = acc01[e] + b1v;
        C[r1 * ldc + cA] = acc10[e] + b0v;
        C[r1 * ldc + cB] = acc11[e] + b1v;
    }
}

// ---------------------------------------------------------------------------
// LSTM gate fusion: z (B x 4U) + bias -> i,f,g,o ; update c ; h -> bf16 (+seq)
// ---------------------------------------------------------------------------
__global__ void lstm_gate_kernel(const float* __restrict__ z, const float* __restrict__ bias,
                                 float* __restrict__ c, bf16* __restrict__ h_bf,
                                 float* __restrict__ seq_f32, bf16* __restrict__ seq_bf,
                                 int t, int S)
{
    int gid = blockIdx.x * blockDim.x + threadIdx.x;     // 0 .. B*U
    int b = gid / UN, u = gid - b * UN;
    const float* zr = z + (long)b * G4;
    float zi = zr[u]          + bias[u];
    float zf = zr[UN + u]     + bias[UN + u];
    float zg = zr[2*UN + u]   + bias[2*UN + u];
    float zo = zr[3*UN + u]   + bias[3*UN + u];
    float si = 1.f / (1.f + expf(-zi));
    float sf = 1.f / (1.f + expf(-zf));
    float so = 1.f / (1.f + expf(-zo));
    float cn = sf * c[gid] + si * tanhf(zg);
    float hv = so * tanhf(cn);
    c[gid]   = cn;
    h_bf[gid] = (bf16)hv;
    long off = ((long)b * S + t) * UN + u;
    if (seq_f32) seq_f32[off] = hv;
    if (seq_bf)  seq_bf[off]  = (bf16)hv;
}

__global__ void zero_state_kernel(float* __restrict__ c, bf16* __restrict__ h_bf, int n)
{
    int i = blockIdx.x * blockDim.x + threadIdx.x;
    if (i < n) { c[i] = 0.f; h_bf[i] = (bf16)0.f; }
}

__global__ void cvt_bf16_kernel(const float* __restrict__ in, bf16* __restrict__ out, long n)
{
    long i = (long)blockIdx.x * blockDim.x + threadIdx.x;
    long stride = (long)gridDim.x * blockDim.x;
    for (; i < n; i += stride) out[i] = (bf16)in[i];
}

__global__ void embed_bf16_kernel(const int* __restrict__ tok, const float* __restrict__ emb,
                                  bf16* __restrict__ out)
{
    int r = blockIdx.x;                 // b*S + s
    long tk = tok[r];
    const float* src = emb + tk * EMBD;
    bf16* dst = out + (long)r * EMBD;
    for (int e = threadIdx.x; e < EMBD; e += blockDim.x) dst[e] = (bf16)src[e];
}

// score[b][s] = sum_u tanh(q[b][u] + kproj[b][s][u]) * wv[u]  (masked)
__global__ void attn_score_kernel(const float* __restrict__ q, const float* __restrict__ kproj,
                                  const float* __restrict__ wv, const int* __restrict__ valid_len,
                                  float* __restrict__ score)
{
    int wtask = blockIdx.x * 8 + (threadIdx.x >> 5);     // one wave per (b,s)
    int b = wtask >> 6;
    int s = wtask & 63;
    int lane = threadIdx.x & 31;
    const float* qb = q + (long)b * UN;
    const float* kp = kproj + ((long)b * SEN + s) * UN;
    float acc = 0.f;
    for (int u = lane; u < UN; u += 32) acc += tanhf(qb[u] + kp[u]) * wv[u];
    #pragma unroll
    for (int off = 16; off > 0; off >>= 1) acc += __shfl_down(acc, off, 32);
    if (lane == 0) score[b * SEN + s] = (s < valid_len[b]) ? acc : -1000000.0f;
}

__global__ void softmax64_kernel(const float* __restrict__ score, float* __restrict__ attn)
{
    __shared__ float sc[SEN];
    int b = blockIdx.x, s = threadIdx.x;
    float v = score[b * SEN + s];
    sc[s] = v;
    __syncthreads();
    float m = -1e30f;
    for (int i = 0; i < SEN; ++i) m = fmaxf(m, sc[i]);
    float sum = 0.f;
    for (int i = 0; i < SEN; ++i) sum += expf(sc[i] - m);
    attn[b * SEN + s] = expf(v - m) / sum;
}

// context[b][u] = sum_s attn[b][s] * en_o[b][s][u]
__global__ void context_kernel(const float* __restrict__ attn, const float* __restrict__ eno,
                               float* __restrict__ ctx, bf16* __restrict__ ctx_bf)
{
    __shared__ float a[SEN];
    int b = blockIdx.x;
    if (threadIdx.x < SEN) a[threadIdx.x] = attn[b * SEN + threadIdx.x];
    __syncthreads();
    for (int u = threadIdx.x; u < UN; u += blockDim.x) {
        float acc = 0.f;
        const float* e0 = eno + (long)b * SEN * UN + u;
        #pragma unroll 4
        for (int s = 0; s < SEN; ++s) acc += a[s] * e0[(long)s * UN];
        ctx[b * UN + u]    = acc;
        ctx_bf[b * UN + u] = (bf16)acc;
    }
}

// ---------------------------------------------------------------------------

extern "C" void kernel_launch(void* const* d_in, const int* in_sizes, int n_in,
                              void* d_out, int out_size, void* d_ws, size_t ws_size,
                              hipStream_t stream)
{
    const int*   en        = (const int*)  d_in[0];
    const int*   fr        = (const int*)  d_in[1];
    const int*   valid_len = (const int*)  d_in[2];
    const float* en_emb    = (const float*)d_in[3];
    const float* de_emb    = (const float*)d_in[4];
    const float* en_W      = (const float*)d_in[5];
    const float* en_U      = (const float*)d_in[6];
    const float* en_b      = (const float*)d_in[7];
    const float* de_W      = (const float*)d_in[8];
    const float* de_U      = (const float*)d_in[9];
    const float* de_b      = (const float*)d_in[10];
    const float* Wq        = (const float*)d_in[11];
    const float* Wk        = (const float*)d_in[12];
    const float* wv        = (const float*)d_in[13];
    const float* Wd        = (const float*)d_in[14];
    const float* bd        = (const float*)d_in[15];
    float* logits = (float*)d_out;

    // --- workspace carve-out (256B aligned) ---
    char* p = (char*)d_ws;
    auto alloc = [&](size_t bytes) -> void* {
        void* r = (void*)p;
        p += (bytes + 255) & ~(size_t)255;
        return r;
    };
    bf16* enW_bf  = (bf16*)alloc((size_t)EMBD * G4 * 2);
    bf16* enU_bf  = (bf16*)alloc((size_t)UN * G4 * 2);
    bf16* deW_bf  = (bf16*)alloc((size_t)(UN + EMBD) * G4 * 2);
    bf16* deU_bf  = (bf16*)alloc((size_t)UN * G4 * 2);
    bf16* Wq_bf   = (bf16*)alloc((size_t)UN * UN * 2);
    bf16* Wk_bf   = (bf16*)alloc((size_t)UN * UN * 2);
    bf16* Wd_bf   = (bf16*)alloc((size_t)UN * VFR * 2);
    bf16* xen_bf  = (bf16*)alloc((size_t)B_ * SEN * EMBD * 2);
    bf16* xfr_bf  = (bf16*)alloc((size_t)B_ * SFR * EMBD * 2);
    bf16* eno_bf  = (bf16*)alloc((size_t)B_ * SEN * UN * 2);
    bf16* hs_bf   = (bf16*)alloc((size_t)B_ * SFR * UN * 2);
    bf16* h_bf    = (bf16*)alloc((size_t)B_ * UN * 2);
    bf16* ctx_bf  = (bf16*)alloc((size_t)B_ * UN * 2);
    float* z      = (float*)alloc((size_t)B_ * G4 * 4);
    float* c      = (float*)alloc((size_t)B_ * UN * 4);
    float* eno    = (float*)alloc((size_t)B_ * SEN * UN * 4);
    float* kproj  = (float*)alloc((size_t)B_ * SEN * UN * 4);
    float* q      = (float*)alloc((size_t)B_ * UN * 4);
    float* score  = (float*)alloc((size_t)B_ * SEN * 4);
    float* attn   = (float*)alloc((size_t)B_ * SEN * 4);
    float* ctx    = (float*)alloc((size_t)B_ * UN * 4);
    (void)ws_size; (void)n_in; (void)in_sizes; (void)out_size;

    // --- weight conversion to bf16 (once per call; L2-resident afterwards) ---
    auto cvt = [&](const float* src, bf16* dst, long n) {
        int blocks = (int)((n + 2047) / 2048); if (blocks > 4096) blocks = 4096;
        cvt_bf16_kernel<<<blocks, 256, 0, stream>>>(src, dst, n);
    };
    cvt(en_W, enW_bf, (long)EMBD * G4);
    cvt(en_U, enU_bf, (long)UN * G4);
    cvt(de_W, deW_bf, (long)(UN + EMBD) * G4);
    cvt(de_U, deU_bf, (long)UN * G4);
    cvt(Wq,   Wq_bf,  (long)UN * UN);
    cvt(Wk,   Wk_bf,  (long)UN * UN);
    cvt(Wd,   Wd_bf,  (long)UN * VFR);

    embed_bf16_kernel<<<B_ * SEN, 256, 0, stream>>>(en, en_emb, xen_bf);
    embed_bf16_kernel<<<B_ * SFR, 256, 0, stream>>>(fr, de_emb, xfr_bf);
    zero_state_kernel<<<(B_ * UN) / 256, 256, 0, stream>>>(c, h_bf, B_ * UN);

    // --- encoder scan: one fused 2-segment WMMA GEMM + gate kernel per step ---
    for (int t = 0; t < SEN; ++t) {
        gemm_bf16_wmma<<<dim3(G4 / 128, B_ / 64), 256, 0, stream>>>(
            xen_bf + (long)t * EMBD, (long)SEN * EMBD, enW_bf, G4, EMBD,
            h_bf, UN, enU_bf, G4, UN,
            nullptr, 0, nullptr, 0, 0,
            z, G4, nullptr);
        lstm_gate_kernel<<<(B_ * UN) / 256, 256, 0, stream>>>(
            z, en_b, c, h_bf, eno, eno_bf, t, SEN);
    }

    // --- kproj = en_o @ Wk  (4096 x 1024 x 1024) ---
    gemm_bf16_wmma<<<dim3(UN / 128, (B_ * SEN) / 64), 256, 0, stream>>>(
        eno_bf, UN, Wk_bf, UN, UN,
        nullptr, 0, nullptr, 0, 0,
        nullptr, 0, nullptr, 0, 0,
        kproj, UN, nullptr);

    // --- decoder scan with attention ---
    for (int t = 0; t < SFR; ++t) {
        // q = h @ Wq
        gemm_bf16_wmma<<<dim3(UN / 128, B_ / 64), 256, 0, stream>>>(
            h_bf, UN, Wq_bf, UN, UN,
            nullptr, 0, nullptr, 0, 0,
            nullptr, 0, nullptr, 0, 0,
            q, UN, nullptr);
        attn_score_kernel<<<(B_ * SEN) / 8, 256, 0, stream>>>(q, kproj, wv, valid_len, score);
        softmax64_kernel<<<B_, SEN, 0, stream>>>(score, attn);
        context_kernel<<<B_, 256, 0, stream>>>(attn, eno, ctx, ctx_bf);
        // z = ctx @ de_W[:U] + x_t @ de_W[U:] + h @ de_U   (3-segment fused GEMM)
        gemm_bf16_wmma<<<dim3(G4 / 128, B_ / 64), 256, 0, stream>>>(
            ctx_bf, UN, deW_bf, G4, UN,
            xfr_bf + (long)t * EMBD, (long)SFR * EMBD, deW_bf + (long)UN * G4, G4, EMBD,
            h_bf, UN, deU_bf, G4, UN,
            z, G4, nullptr);
        lstm_gate_kernel<<<(B_ * UN) / 256, 256, 0, stream>>>(
            z, de_b, c, h_bf, nullptr, hs_bf, t, SFR);
    }

    // --- logits = hs @ Wd + bd  (4096 x 1024 x 16000) ---
    gemm_bf16_wmma<<<dim3(VFR / 128, (B_ * SFR) / 64), 256, 0, stream>>>(
        hs_bf, UN, Wd_bf, VFR, UN,
        nullptr, 0, nullptr, 0, 0,
        nullptr, 0, nullptr, 0, 0,
        logits, VFR, bd);
}
